// WindowMSA_1185410974381
// MI455X (gfx1250) — compile-verified, matrix operand order checked by reading, not run on hardware
//
#include <hip/hip_runtime.h>
#include <hip/hip_bf16.h>

// ---------------------------------------------------------------------------
// Window-MSA (Swin) for MI455X / gfx1250, wave32 + v_wmma_f32_16x16x32_bf16.
//
//   B = 4096 windows, N = 49 tokens (padded to 64), C = 256, H = 8, hd = 32
//   M = B*49 = 200704 rows (divisible by 16 -> 12544 M-tiles)
//
// Pipeline (bf16 tensor math, fp32 accumulate):
//   K0: convert x, qkv_w, proj_w to bf16 once (kills per-wave cvt VALU work;
//       x_bf16 = 103 MB -> L2-resident across K1's N-group re-reads)
//   K1: qkv  = x @ qkv_w^T + qkv_b   (q pre-scaled by hd^-0.5)  -> ws (bf16)
//   K2: per (b,h): S = q k^T + relbias + mask; softmax; O = P v -> ws (bf16)
//   K3: out  = O @ proj_w^T + proj_b                            -> d_out (f32)
// ---------------------------------------------------------------------------

typedef __attribute__((ext_vector_type(16))) __bf16 v16bf;
typedef __attribute__((ext_vector_type(8)))  float  v8f;

#define NTOK   49
#define NPAD   64
#define EMBED  256
#define HEADS  8
#define HD     32
#define NWIN   64
#define MROWS  200704            // 4096 * 49
#define MTILES 12544             // MROWS / 16
#define PLDS_LD 72               // padded LDS row stride (bf16 elems)

__device__ __forceinline__ int lane_id() { return threadIdx.x & 31; }

__device__ __forceinline__ v8f wmma_bf16(v16bf a, v16bf b, v8f c) {
  // (neg_a, A, neg_b, B, c_mod, C, reuse_a, reuse_b)
  return __builtin_amdgcn_wmma_f32_16x16x32_bf16(false, a, false, b,
                                                 (short)0, c, false, false);
}

// 16x32 bf16 A/B fragment from a row-major bf16 matrix.
// Lane L supplies matrix row (r0 + (L&15)); element i maps to column
//   k0 + ((L>>4)*8) + (i&7) + ((i>>3)*16)     [ISA 7.12.2, 16-bit A layout]
// Each 8-element half is 16 contiguous bytes -> one global_load_b128.
__device__ __forceinline__ v16bf frag_bf16(const __bf16* __restrict__ p, int ld,
                                           int r0, int k0) {
  const int L = lane_id();
  const __bf16* row = p + (long)(r0 + (L & 15)) * ld + k0 + ((L >> 4) << 3);
  v16bf f;
#pragma unroll
  for (int i = 0; i < 8; ++i) f[i]     = row[i];
#pragma unroll
  for (int i = 0; i < 8; ++i) f[i + 8] = row[i + 16];
  return f;
}

// q/k head fragment: lane row = token (zero-padded past 49), cols = head dims.
__device__ __forceinline__ v16bf frag_head(const __bf16* __restrict__ qkv,
                                           long rowbase, int t0, int off) {
  const int L = lane_id();
  const int t = t0 + (L & 15);
  const bool ok = t < NTOK;
  const __bf16* row = qkv + (rowbase + t) * (3L * EMBED) + off + ((L >> 4) << 3);
  const __bf16 z = (__bf16)0.0f;
  v16bf f;
#pragma unroll
  for (int i = 0; i < 8; ++i) f[i]     = ok ? row[i]      : z;
#pragma unroll
  for (int i = 0; i < 8; ++i) f[i + 8] = ok ? row[i + 16] : z;
  return f;
}

// V as B-operand: lane = output column d, element i = token K (zero-padded).
__device__ __forceinline__ v16bf frag_v(const __bf16* __restrict__ qkv,
                                        long rowbase, int hoff, int d0, int kk) {
  const int L = lane_id();
  const int d  = d0 + (L & 15);
  const int tb = kk + ((L >> 4) << 3);
  const __bf16 z = (__bf16)0.0f;
  v16bf f;
#pragma unroll
  for (int i = 0; i < 16; ++i) {
    const int t = tb + (i & 7) + ((i >> 3) << 4);
    f[i] = (t < NTOK) ? qkv[(rowbase + t) * (3L * EMBED) + hoff + d] : z;
  }
  return f;
}

// P A-fragment out of this wave's LDS staging tile.
__device__ __forceinline__ v16bf frag_p(const __bf16* __restrict__ pl, int kk) {
  const int L = lane_id();
  const __bf16* row = pl + (L & 15) * PLDS_LD + kk + ((L >> 4) << 3);
  v16bf f;
#pragma unroll
  for (int i = 0; i < 8; ++i) f[i]     = row[i];
#pragma unroll
  for (int i = 0; i < 8; ++i) f[i + 8] = row[i + 16];
  return f;
}

// ---------------------------------------------------------------------------
// K0: one-shot fp32 -> bf16 conversion (n must be a multiple of 4).
// ---------------------------------------------------------------------------
__global__ void __launch_bounds__(256)
cvt_kernel(const float* __restrict__ src, __bf16* __restrict__ dst, long n) {
  const long i = ((long)blockIdx.x * blockDim.x + threadIdx.x) * 4;
  if (i < n) {
    const float4 v = *(const float4*)(src + i);
    dst[i + 0] = (__bf16)v.x;
    dst[i + 1] = (__bf16)v.y;
    dst[i + 2] = (__bf16)v.z;
    dst[i + 3] = (__bf16)v.w;
  }
}

// ---------------------------------------------------------------------------
// K1: QKV projection. 4 waves/block, each wave -> one 16x128 output tile.
//   grid = (MTILES/4, 768/128), block = 128
// ---------------------------------------------------------------------------
__global__ void __launch_bounds__(128)
qkv_kernel(const __bf16* __restrict__ x, const __bf16* __restrict__ qkv_w,
           const float* __restrict__ qkv_b, __bf16* __restrict__ qkv_out) {
  const int wid = threadIdx.x >> 5;
  const int L   = threadIdx.x & 31;
  const int m0  = (blockIdx.x * 4 + wid) * 16;
  const int n0  = blockIdx.y * 128;

  v8f acc[8];
#pragma unroll
  for (int j = 0; j < 8; ++j)
#pragma unroll
    for (int i = 0; i < 8; ++i) acc[j][i] = 0.0f;

#pragma unroll
  for (int k0 = 0; k0 < EMBED; k0 += 32) {
    const v16bf a = frag_bf16(x, EMBED, m0, k0);             // A reused 8x
#pragma unroll
    for (int j = 0; j < 8; ++j) {
      const v16bf b = frag_bf16(qkv_w, EMBED, n0 + j * 16, k0); // B[K,N]=W[N,K]
      acc[j] = wmma_bf16(a, b, acc[j]);
    }
  }

  const float scale = 0.17677669529663687f;                  // hd^-0.5
#pragma unroll
  for (int j = 0; j < 8; ++j) {
    const int n   = n0 + j * 16 + (L & 15);
    const float bs = qkv_b[n];
    const float sc = (n < EMBED) ? scale : 1.0f;             // pre-scale q only
#pragma unroll
    for (int r = 0; r < 8; ++r) {
      const int m = m0 + r + ((L >> 4) << 3);
      qkv_out[(long)m * (3 * EMBED) + n] = (__bf16)((acc[j][r] + bs) * sc);
    }
  }
}

// ---------------------------------------------------------------------------
// K2: fused attention. block = (window b, head h), 4 waves = 4 query tiles.
//   grid = (4096, 8), block = 128
// ---------------------------------------------------------------------------
__global__ void __launch_bounds__(128)
attn_kernel(const __bf16* __restrict__ qkv, const float* __restrict__ mask,
            const float* __restrict__ bias_table, __bf16* __restrict__ attn_out) {
  __shared__ __bf16 Plds[4][16 * PLDS_LD];

  const int b   = blockIdx.x;
  const int h   = blockIdx.y;
  const int wid = threadIdx.x >> 5;
  const int L   = threadIdx.x & 31;
  const int q0  = wid * 16;
  const long rowbase = (long)b * NTOK;
  const int hoff_q = h * HD;
  const int hoff_k = EMBED + h * HD;
  const int hoff_v = 2 * EMBED + h * HD;

  // ---- S = q k^T (q pre-scaled in K1), 64x64 padded, this wave: 16x64 ----
  const v16bf qf = frag_head(qkv, rowbase, q0, hoff_q);
  v8f s[4];
#pragma unroll
  for (int t = 0; t < 4; ++t) {
#pragma unroll
    for (int i = 0; i < 8; ++i) s[t][i] = 0.0f;
    const v16bf kf = frag_head(qkv, rowbase, t * 16, hoff_k);
    s[t] = wmma_bf16(qf, kf, s[t]);
  }

  // ---- + relative position bias (computed analytically) + window mask ----
  const int w    = b & (NWIN - 1);
  const int ncol = L & 15;
  const int mrb  = q0 + ((L >> 4) << 3);
#pragma unroll
  for (int t = 0; t < 4; ++t) {
    const int n = t * 16 + ncol;
#pragma unroll
    for (int r = 0; r < 8; ++r) {
      const int m = mrb + r;
      if (n >= NTOK) {
        s[t][r] = -1e30f;                    // key padding -> exp() == 0
      } else if (m < NTOK) {
        const int di = (m / 7) - (n / 7) + 6;
        const int dj = (m % 7) - (n % 7) + 6;
        s[t][r] += bias_table[(di * 13 + dj) * HEADS + h]
                 + mask[((long)w * NTOK + m) * NTOK + n];
      }
    }
  }

  // ---- softmax across N: rows live in 16-lane half-wave groups ----
#pragma unroll
  for (int r = 0; r < 8; ++r) {
    float mv = fmaxf(fmaxf(s[0][r], s[1][r]), fmaxf(s[2][r], s[3][r]));
#pragma unroll
    for (int o = 8; o >= 1; o >>= 1) mv = fmaxf(mv, __shfl_xor(mv, o, 16));
    float sum = 0.0f;
#pragma unroll
    for (int t = 0; t < 4; ++t) {
      const float e = __expf(s[t][r] - mv);
      s[t][r] = e;
      sum += e;
    }
#pragma unroll
    for (int o = 8; o >= 1; o >>= 1) sum += __shfl_xor(sum, o, 16);
    const float inv = 1.0f / sum;
#pragma unroll
    for (int t = 0; t < 4; ++t) s[t][r] *= inv;
  }

  // ---- C-layout -> A-fragment layout via LDS (per-wave tile) ----
  __bf16* pl = Plds[wid];
#pragma unroll
  for (int t = 0; t < 4; ++t)
#pragma unroll
    for (int r = 0; r < 8; ++r)
      pl[(r + ((L >> 4) << 3)) * PLDS_LD + t * 16 + ncol] = (__bf16)s[t][r];
  __syncthreads();

  // ---- O = P @ V : 16x32 output, K = 64 in two 32-chunks ----
  v8f o0, o1;
#pragma unroll
  for (int i = 0; i < 8; ++i) { o0[i] = 0.0f; o1[i] = 0.0f; }
#pragma unroll
  for (int kk = 0; kk < NPAD; kk += 32) {
    const v16bf pf = frag_p(pl, kk);
    const v16bf v0 = frag_v(qkv, rowbase, hoff_v, 0,  kk);
    const v16bf v1 = frag_v(qkv, rowbase, hoff_v, 16, kk);
    o0 = wmma_bf16(pf, v0, o0);
    o1 = wmma_bf16(pf, v1, o1);
  }

  // ---- store O (bf16) back as [B*49, 256] head-interleaved ----
#pragma unroll
  for (int r = 0; r < 8; ++r) {
    const int tkn = q0 + r + ((L >> 4) << 3);
    if (tkn < NTOK) {
      const long orow = (rowbase + tkn) * EMBED;
      attn_out[orow + h * HD + ncol]      = (__bf16)o0[r];
      attn_out[orow + h * HD + 16 + ncol] = (__bf16)o1[r];
    }
  }
}

// ---------------------------------------------------------------------------
// K3: output projection. 4 waves/block, wave -> 16x128 tile, fp32 out + bias.
//   grid = (MTILES/4, 256/128), block = 128
// ---------------------------------------------------------------------------
__global__ void __launch_bounds__(128)
proj_kernel(const __bf16* __restrict__ a_in, const __bf16* __restrict__ proj_w,
            const float* __restrict__ proj_b, float* __restrict__ out) {
  const int wid = threadIdx.x >> 5;
  const int L   = threadIdx.x & 31;
  const int m0  = (blockIdx.x * 4 + wid) * 16;
  const int n0  = blockIdx.y * 128;

  v8f acc[8];
#pragma unroll
  for (int j = 0; j < 8; ++j)
#pragma unroll
    for (int i = 0; i < 8; ++i) acc[j][i] = 0.0f;

#pragma unroll
  for (int k0 = 0; k0 < EMBED; k0 += 32) {
    const v16bf a = frag_bf16(a_in, EMBED, m0, k0);
#pragma unroll
    for (int j = 0; j < 8; ++j) {
      const v16bf b = frag_bf16(proj_w, EMBED, n0 + j * 16, k0);
      acc[j] = wmma_bf16(a, b, acc[j]);
    }
  }

#pragma unroll
  for (int j = 0; j < 8; ++j) {
    const int n = n0 + j * 16 + (L & 15);
    const float bs = proj_b[n];
#pragma unroll
    for (int r = 0; r < 8; ++r) {
      const int m = m0 + r + ((L >> 4) << 3);
      out[(long)m * EMBED + n] = acc[j][r] + bs;
    }
  }
}

// ---------------------------------------------------------------------------
extern "C" void kernel_launch(void* const* d_in, const int* in_sizes, int n_in,
                              void* d_out, int out_size, void* d_ws, size_t ws_size,
                              hipStream_t stream) {
  const float* x          = (const float*)d_in[0];  // [4096,49,256]
  const float* mask       = (const float*)d_in[1];  // [64,49,49]
  const float* qkv_w      = (const float*)d_in[2];  // [768,256]
  const float* qkv_b      = (const float*)d_in[3];  // [768]
  const float* proj_w     = (const float*)d_in[4];  // [256,256]
  const float* proj_b     = (const float*)d_in[5];  // [256]
  const float* bias_table = (const float*)d_in[6];  // [169,8]
  float* out = (float*)d_out;                       // [4096,49,256]

  // workspace (bf16): qkv [200704,768] | attn [200704,256] | x [200704,256]
  //                   | qkv_w [768,256] | proj_w [256,256]
  __bf16* ws_qkv  = (__bf16*)d_ws;
  __bf16* ws_attn = ws_qkv  + (size_t)MROWS * (3 * EMBED);
  __bf16* ws_x    = ws_attn + (size_t)MROWS * EMBED;
  __bf16* ws_wq   = ws_x    + (size_t)MROWS * EMBED;
  __bf16* ws_wp   = ws_wq   + (size_t)(3 * EMBED) * EMBED;

  // K0: one-shot conversions (x read once; bf16 x is L2-resident afterwards)
  {
    const long nx = (long)MROWS * EMBED;
    cvt_kernel<<<(int)((nx / 4 + 255) / 256), 256, 0, stream>>>(x, ws_x, nx);
    const long nwq = (long)(3 * EMBED) * EMBED;
    cvt_kernel<<<(int)((nwq / 4 + 255) / 256), 256, 0, stream>>>(qkv_w, ws_wq, nwq);
    const long nwp = (long)EMBED * EMBED;
    cvt_kernel<<<(int)((nwp / 4 + 255) / 256), 256, 0, stream>>>(proj_w, ws_wp, nwp);
  }

  dim3 gA(MTILES / 4, (3 * EMBED) / 128);
  qkv_kernel<<<gA, 128, 0, stream>>>(ws_x, ws_wq, qkv_b, ws_qkv);

  dim3 gB(4096, HEADS);
  attn_kernel<<<gB, 128, 0, stream>>>(ws_qkv, mask, bias_table, ws_attn);

  dim3 gC(MTILES / 4, EMBED / 128);
  proj_kernel<<<gC, 128, 0, stream>>>(ws_attn, ws_wp, proj_b, out);
}
